// SudokuNN_81174881894715
// MI455X (gfx1250) — matrix operand order repaired and checked by reading, not compiled, not run on hardware
//
#include <hip/hip_runtime.h>

#define NTOT   5184      // 64 boards * 81 cells
#define HIDN   96
#define EDGES  103680    // 1620 * 64
#define NSTEP  8
#define APAD   104       // LDS activation row stride in halves (16B aligned)

typedef __attribute__((ext_vector_type(16))) _Float16 v16h;
typedef __attribute__((ext_vector_type(8)))  _Float16 v8h;
typedef __attribute__((ext_vector_type(8)))  float    v8f;
typedef int g4i __attribute__((vector_size(16)));   // matches builtin's V4i

#define AS_GLOBAL __attribute__((address_space(1)))
#define AS_SHARED __attribute__((address_space(3)))

// ---------------------------------------------------------------------------
// A-fragment loader for V_WMMA_F32_16X16X32_F16.
// ISA 7.12.2 (16-bit A 16x32): lane L holds row M = L&15; halves 0..7 hold
// K = hi8..hi8+7 and halves 8..15 hold K = hi8+16..hi8+23 with hi8 = 8*(L>=16).
// So a fragment is two contiguous 16B loads from a row-major f16 row pointer.
// ---------------------------------------------------------------------------
static __device__ __forceinline__ v16h load_a_frag(const _Float16* p, int lane) {
  const _Float16* q = p + ((lane >> 4) << 3);
  v8h lo = *(const v8h*)(q);
  v8h hi = *(const v8h*)(q + 16);
  v16h a;
#pragma unroll
  for (int i = 0; i < 8; ++i) { a[i] = lo[i]; a[i + 8] = hi[i]; }
  return a;
}

// Same fragment, but sourced from an f32 row (converts on the fly).
static __device__ __forceinline__ v16h load_a_frag_f32(const float* p, int lane) {
  const float* q = p + ((lane >> 4) << 3);
  v8f lo = *(const v8f*)(q);
  v8f hi = *(const v8f*)(q + 16);
  v16h a;
#pragma unroll
  for (int i = 0; i < 8; ++i) {
    a[i] = (_Float16)lo[i];
    a[i + 8] = (_Float16)hi[i];
  }
  return a;
}

static __device__ __forceinline__ v8f wmma_f16(v16h a, v16h b, v8f c) {
  return __builtin_amdgcn_wmma_f32_16x16x32_f16(false, a, false, b, (short)0, c,
                                                false, false);
}

static __device__ __forceinline__ float sigmoidf_(float x) {
  return 1.0f / (1.0f + __expf(-x));
}

// ---------------------------------------------------------------------------
// Weight packing: B-fragment order. Fragment (kt,nt) occupies 512 halves:
// half index = lane*16 + i, from W[kt*32 + 16*(i>=8) + 8*(lane>=16) + (i&7)][nt*16 + (lane&15)].
// Hot kernels then load one 32B v16h per lane per fragment.
// ---------------------------------------------------------------------------
__global__ void pack_b_kernel(const float* __restrict__ W, int fi, int fo,
                              _Float16* __restrict__ out) {
  int tid = blockIdx.x * blockDim.x + threadIdx.x;
  int NT = fo >> 4;
  int total = (fi >> 5) * NT * 512;
  if (tid >= total) return;
  int f = tid >> 9, rem = tid & 511;
  int lane = rem >> 4, i = rem & 15;
  int kt = f / NT, nt = f - kt * NT;
  int K = kt * 32 + ((i >> 3) << 4) + ((lane >> 4) << 3) + (i & 7);
  int N = nt * 16 + (lane & 15);
  out[tid] = (_Float16)W[(size_t)K * fo + N];
}

// Stacked LSTM weights: rows 0..191 = W_ih (192x384), rows 192..287 = W_hh (96x384)
__global__ void pack_lstm_kernel(const float* __restrict__ Wih,
                                 const float* __restrict__ Whh,
                                 _Float16* __restrict__ out) {
  int tid = blockIdx.x * blockDim.x + threadIdx.x;
  const int NT = 24;                       // 384/16
  int total = 9 * NT * 512;                // 288/32 k-tiles
  if (tid >= total) return;
  int f = tid >> 9, rem = tid & 511;
  int lane = rem >> 4, i = rem & 15;
  int kt = f / NT, nt = f - kt * NT;
  int K = kt * 32 + ((i >> 3) << 4) + ((lane >> 4) << 3) + (i & 7);
  int N = nt * 16 + (lane & 15);
  float v = (K < 192) ? Wih[(size_t)K * 384 + N] : Whh[(size_t)(K - 192) * 384 + N];
  out[tid] = (_Float16)v;
}

// ---------------------------------------------------------------------------
// Input encoder: feats = [cost@digit_embed | row_embed | col_embed] -> 4-layer MLP.
// One block (96 threads) per node; ~2% of FLOPs, VALU is fine.
// ---------------------------------------------------------------------------
__global__ __launch_bounds__(96) void input_mlp_kernel(
    const float* __restrict__ cost, const int* __restrict__ rows,
    const int* __restrict__ cols, const float* __restrict__ DE,
    const float* __restrict__ RE, const float* __restrict__ CE,
    const float* __restrict__ W0, const float* __restrict__ b0,
    const float* __restrict__ W1, const float* __restrict__ b1,
    const float* __restrict__ W2, const float* __restrict__ b2,
    const float* __restrict__ W3, const float* __restrict__ b3,
    _Float16* __restrict__ x16, _Float16* __restrict__ h16) {
  __shared__ float sa[96];
  __shared__ float sb[96];
  int n = blockIdx.x;
  int j = threadIdx.x;
  if (j < 48) {
    float v;
    if (j < 16) {
      v = 0.0f;
#pragma unroll
      for (int d = 0; d < 10; ++d) v += cost[n * 10 + d] * DE[d * 16 + j];
    } else if (j < 32) {
      v = RE[rows[n] * 16 + (j - 16)];
    } else {
      v = CE[cols[n] * 16 + (j - 32)];
    }
    sa[j] = v;
  }
  __syncthreads();
  float a = b0[j];
  for (int k = 0; k < 48; ++k) a += sa[k] * W0[k * 96 + j];
  a = fmaxf(a, 0.0f);
  sb[j] = a;
  __syncthreads();
  a = b1[j];
  for (int k = 0; k < 96; ++k) a += sb[k] * W1[k * 96 + j];
  a = fmaxf(a, 0.0f);
  sa[j] = a;
  __syncthreads();
  a = b2[j];
  for (int k = 0; k < 96; ++k) a += sa[k] * W2[k * 96 + j];
  a = fmaxf(a, 0.0f);
  sb[j] = a;
  __syncthreads();
  a = b3[j];
  for (int k = 0; k < 96; ++k) a += sb[k] * W3[k * 96 + j];
  x16[(size_t)n * HIDN + j] = (_Float16)a;   // x (gate input, constant)
  h16[(size_t)n * HIDN + j] = (_Float16)a;   // h0 = x (message input)
}

// ---------------------------------------------------------------------------
// Edge message MLP, fully fused: gather h[src]/h[dst] -> 192->96->96->96->96
// with WMMA f32_16x16x32_f16 -> scatter-add into m (segment_sum by dst).
// Block = 256 threads (8 wave32) handles 64 edges x 96 outputs.
// Wave w: mt = w>>1 (16-edge row tile), ng = w&1 (3 of 6 col tiles).
// W1+W2 fragments (36 KB, contiguous) are staged into LDS with async
// global->LDS copies issued in the prologue; the copy overlaps the whole
// layer-0 phase and is waited on (ASYNCcnt) right before the layer-1 barrier.
// ---------------------------------------------------------------------------
__global__ __launch_bounds__(256) void msg_mlp_kernel(
    const _Float16* __restrict__ h16, const int* __restrict__ src,
    const int* __restrict__ dst, const _Float16* __restrict__ pW0,
    const _Float16* __restrict__ pW12,   // W1 frags [0..17], W2 frags [18..35]
    const _Float16* __restrict__ pW3, const float* __restrict__ b0,
    const float* __restrict__ b1, const float* __restrict__ b2,
    const float* __restrict__ b3, float* __restrict__ m32) {
  __shared__ int sSrc[64];
  __shared__ int sDst[64];
  __shared__ _Float16 act[2][64][APAD];          // 26624 B
  __shared__ _Float16 wlds[36 * 512];            // 36864 B (total LDS 64000 B)

  const int tbase = blockIdx.x * 64;
  const int tid = threadIdx.x;
  if (tid < 64) {
    sSrc[tid] = src[tbase + tid];
    sDst[tid] = dst[tbase + tid];
  }

  // ---- kick off async staging of W1,W2 fragments into LDS ----
  // 36864 B = 2304 16-byte chunks; 256 threads x 9 iterations.
#pragma unroll
  for (int k = 0; k < 9; ++k) {
    int c16 = tid + k * 256;
    const _Float16* gp = pW12 + c16 * 8;
    _Float16* lp = &wlds[c16 * 8];
#if __has_builtin(__builtin_amdgcn_global_load_async_to_lds_b128)
    __builtin_amdgcn_global_load_async_to_lds_b128(
        (AS_GLOBAL g4i*)(uintptr_t)gp, (AS_SHARED g4i*)(uintptr_t)lp, 0, 0);
#else
    *(v8h*)lp = *(const v8h*)gp;   // fallback: plain load + ds_store
#endif
  }
  __syncthreads();   // sSrc/sDst visible

  const int w = tid >> 5, lane = tid & 31;
  const int mt = w >> 1, ng = w & 1;
  const int arow = lane & 15;              // A-fragment row (both half-lanes)
  const int crowoff = (lane >> 4) << 3;    // C-fragment row offset (0 or 8)
  const int ccol = lane & 15;              // C-fragment col within tile

  // ---- layer 0: K = 192 (K<96 -> h[src], K>=96 -> h[dst]), B from global ----
  v8f acc[3] = {};
  const int eA = mt * 16 + arow;
#pragma unroll
  for (int c = 0; c < 6; ++c) {
    int node = (c < 3) ? sSrc[eA] : sDst[eA];
    int k0 = (c < 3 ? c : c - 3) * 32;
    v16h a = load_a_frag(h16 + (size_t)node * HIDN + k0, lane);
#pragma unroll
    for (int j = 0; j < 3; ++j) {
      int nt = ng * 3 + j;
      v16h b = *(const v16h*)(pW0 + (((c * 6 + nt) << 9) + (lane << 4)));
      acc[j] = wmma_f16(a, b, acc[j]);
    }
  }
#pragma unroll
  for (int j = 0; j < 3; ++j) {
    int col = (ng * 3 + j) * 16 + ccol;
    float bb = b0[col];
#pragma unroll
    for (int r = 0; r < 8; ++r) {
      float v = acc[j][r] + bb;
      act[0][mt * 16 + r + crowoff][col] = (_Float16)fmaxf(v, 0.0f);
    }
  }
  // weights staged async must be resident before anyone reads them
#if __has_builtin(__builtin_amdgcn_global_load_async_to_lds_b128)
#if __has_builtin(__builtin_amdgcn_s_wait_asynccnt)
  __builtin_amdgcn_s_wait_asynccnt(0);
#else
  asm volatile("s_wait_asynccnt 0x0" ::: "memory");
#endif
#endif
  __syncthreads();

  // ---- hidden layers 1,2 (K = 96, A in LDS ping-pong, B from LDS) ----
  const float* bmid[2] = {b1, b2};
#pragma unroll
  for (int l = 0; l < 2; ++l) {
    int ib = l & 1, ob = ib ^ 1;
    v8f ac[3] = {};
#pragma unroll
    for (int c = 0; c < 3; ++c) {
      v16h a = load_a_frag(&act[ib][mt * 16 + arow][c * 32], lane);
#pragma unroll
      for (int j = 0; j < 3; ++j) {
        int nt = ng * 3 + j;
        v16h b = *(const v16h*)(&wlds[((l * 18 + c * 6 + nt) << 9) + (lane << 4)]);
        ac[j] = wmma_f16(a, b, ac[j]);
      }
    }
#pragma unroll
    for (int j = 0; j < 3; ++j) {
      int col = (ng * 3 + j) * 16 + ccol;
      float bb = bmid[l][col];
#pragma unroll
      for (int r = 0; r < 8; ++r) {
        float v = ac[j][r] + bb;
        act[ob][mt * 16 + r + crowoff][col] = (_Float16)fmaxf(v, 0.0f);
      }
    }
    __syncthreads();
  }

  // ---- layer 3 (K = 96, no relu, B from global) + scatter-add to m[dst] ----
  v8f a3[3] = {};
#pragma unroll
  for (int c = 0; c < 3; ++c) {
    v16h a = load_a_frag(&act[0][mt * 16 + arow][c * 32], lane);
#pragma unroll
    for (int j = 0; j < 3; ++j) {
      int nt = ng * 3 + j;
      v16h b = *(const v16h*)(pW3 + (((c * 6 + nt) << 9) + (lane << 4)));
      a3[j] = wmma_f16(a, b, a3[j]);
    }
  }
#pragma unroll
  for (int j = 0; j < 3; ++j) {
    int col = (ng * 3 + j) * 16 + ccol;
    float bb = b3[col];
#pragma unroll
    for (int r = 0; r < 8; ++r) {
      int eloc = mt * 16 + r + crowoff;
      int node = sDst[eloc];
      unsafeAtomicAdd(m32 + (size_t)node * HIDN + col, a3[j][r] + bb);
    }
  }
}

// ---------------------------------------------------------------------------
// LSTM gate GEMM: gates = [x | m | h_rnn] (288) @ [W_ih ; W_hh] (288x384).
// m is consumed directly from the f32 atomic accumulator (converted while
// building the A fragment) -- no separate conversion pass.
// Block = 256 threads / 8 waves handles 64 nodes x 384 cols.
// ---------------------------------------------------------------------------
__global__ __launch_bounds__(256) void gates_gemm_kernel(
    const _Float16* __restrict__ x16, const float* __restrict__ m32,
    const _Float16* __restrict__ hr16, const _Float16* __restrict__ pG,
    float* __restrict__ gates) {
  const int tid = threadIdx.x;
  const int w = tid >> 5, lane = tid & 31;
  const int mt = w >> 1, nh = w & 1;
  const int arow = lane & 15;
  const int node0 = blockIdx.x * 64 + mt * 16;
  const size_t rowoff = (size_t)(node0 + arow) * HIDN;

  v8f acc[12] = {};
#pragma unroll
  for (int c = 0; c < 9; ++c) {
    int k0 = (c % 3) * 32;
    v16h a;
    if (c < 3)
      a = load_a_frag(x16 + rowoff + k0, lane);
    else if (c < 6)
      a = load_a_frag_f32(m32 + rowoff + k0, lane);
    else
      a = load_a_frag(hr16 + rowoff + k0, lane);
#pragma unroll
    for (int j = 0; j < 12; ++j) {
      int nt = nh * 12 + j;
      v16h b = *(const v16h*)(pG + (((c * 24 + nt) << 9) + (lane << 4)));
      acc[j] = wmma_f16(a, b, acc[j]);
    }
  }
#pragma unroll
  for (int j = 0; j < 12; ++j) {
    int col = (nh * 12 + j) * 16 + (lane & 15);
#pragma unroll
    for (int r = 0; r < 8; ++r) {
      int node = node0 + r + ((lane >> 4) << 3);
      gates[(size_t)node * 384 + col] = acc[j][r];
    }
  }
}

// Pointwise LSTM cell update (gate order i,f,g,o); also re-zeros the m
// accumulator for the next step's scatter-add.
__global__ void lstm_update_kernel(const float* __restrict__ gates,
                                   float* __restrict__ c32,
                                   float* __restrict__ hs32,
                                   _Float16* __restrict__ h16,
                                   _Float16* __restrict__ hr16,
                                   float* __restrict__ m32, int t) {
  int idx = blockIdx.x * blockDim.x + threadIdx.x;  // node*96 + j
  int n = idx / HIDN, j = idx - n * HIDN;
  const float* g = gates + (size_t)n * 384;
  float gi = g[j], gf = g[96 + j], gg = g[192 + j], go = g[288 + j];
  float c = c32[idx];
  float nc = sigmoidf_(gf) * c + sigmoidf_(gi) * tanhf(gg);
  float nh = sigmoidf_(go) * tanhf(nc);
  c32[idx] = nc;
  hs32[(size_t)t * (NTOT * HIDN) + idx] = nh;
  h16[idx] = (_Float16)nh;
  hr16[idx] = (_Float16)nh;
  m32[idx] = 0.0f;
}

__global__ void zero_loss_kernel(float* p) { *p = 0.0f; }

// Output head: logits = h @ W_out + b; preds = argmax; loss via log-softmax.
__global__ void output_head_kernel(const float* __restrict__ hs32,
                                   const float* __restrict__ Wout,
                                   const float* __restrict__ bout,
                                   const int* __restrict__ labels,
                                   float* __restrict__ preds,
                                   float* __restrict__ loss, float invCount) {
  int idx = blockIdx.x * blockDim.x + threadIdx.x;  // t*NTOT + n
  const float* h = hs32 + (size_t)idx * HIDN;
  float lg[10];
#pragma unroll
  for (int d = 0; d < 10; ++d) lg[d] = bout[d];
  for (int k = 0; k < HIDN; ++k) {
    float hv = h[k];
#pragma unroll
    for (int d = 0; d < 10; ++d) lg[d] += hv * Wout[k * 10 + d];
  }
  int best = 0;
  float bv = lg[0];
#pragma unroll
  for (int d = 1; d < 10; ++d)
    if (lg[d] > bv) { bv = lg[d]; best = d; }
  float s = 0.0f;
#pragma unroll
  for (int d = 0; d < 10; ++d) s += __expf(lg[d] - bv);
  float lse = bv + __logf(s);
  int lab = labels[idx % NTOT];
  preds[idx] = (float)best;
  unsafeAtomicAdd(loss, (lse - lg[lab]) * invCount);
}

// ---------------------------------------------------------------------------
extern "C" void kernel_launch(void* const* d_in, const int* in_sizes, int n_in,
                              void* d_out, int out_size, void* d_ws,
                              size_t ws_size, hipStream_t stream) {
  // setup_inputs() dict order (params flattened in insertion order)
  const float* cost = (const float*)d_in[0];
  const int* rows = (const int*)d_in[1];
  const int* cols = (const int*)d_in[2];
  const int* labels = (const int*)d_in[3];
  const int* src = (const int*)d_in[4];
  const int* dst = (const int*)d_in[5];
  const float* DE = (const float*)d_in[6];   // digit_embed [10,16]
  const float* RE = (const float*)d_in[7];   // row_embed   [9,16]
  const float* CE = (const float*)d_in[8];   // col_embed   [9,16]
  const float* iW0 = (const float*)d_in[9],  *ib0 = (const float*)d_in[10];
  const float* iW1 = (const float*)d_in[11], *ib1 = (const float*)d_in[12];
  const float* iW2 = (const float*)d_in[13], *ib2 = (const float*)d_in[14];
  const float* iW3 = (const float*)d_in[15], *ib3 = (const float*)d_in[16];
  const float* mW0 = (const float*)d_in[17], *mb0 = (const float*)d_in[18];
  const float* mW1 = (const float*)d_in[19], *mb1 = (const float*)d_in[20];
  const float* mW2 = (const float*)d_in[21], *mb2 = (const float*)d_in[22];
  const float* mW3 = (const float*)d_in[23], *mb3 = (const float*)d_in[24];
  const float* Wih = (const float*)d_in[25];
  const float* Whh = (const float*)d_in[26];
  const float* Wout = (const float*)d_in[27];
  const float* bout = (const float*)d_in[28];

  // workspace carve-up (256B aligned)
  char* ws = (char*)d_ws;
  size_t off = 0;
  auto take = [&](size_t bytes) -> char* {
    char* p = ws + off;
    off = (off + bytes + 255) & ~(size_t)255;
    return p;
  };
  _Float16* pMW0  = (_Float16*)take(6 * 6 * 512 * 2);    // msg W0 packed
  _Float16* pMW12 = (_Float16*)take(36 * 512 * 2);       // msg W1+W2 packed, contiguous
  _Float16* pMW3  = (_Float16*)take(3 * 6 * 512 * 2);
  _Float16* pG    = (_Float16*)take(9 * 24 * 512 * 2);   // [Wih;Whh] packed
  const size_t NH = (size_t)NTOT * HIDN;
  _Float16* x16  = (_Float16*)take(NH * 2);
  _Float16* h16  = (_Float16*)take(NH * 2);
  _Float16* hr16 = (_Float16*)take(NH * 2);
  float* m32     = (float*)take(NH * 4);
  float* c32     = (float*)take(NH * 4);
  float* gates   = (float*)take((size_t)NTOT * 384 * 4);
  float* hs32    = (float*)take((size_t)NSTEP * NH * 4);
  (void)ws_size; (void)in_sizes; (void)n_in;

  // one-time weight packing (deterministic, re-run every call)
  pack_b_kernel<<<(6 * 6 * 512 + 255) / 256, 256, 0, stream>>>(mW0, 192, 96, pMW0);
  pack_b_kernel<<<(3 * 6 * 512 + 255) / 256, 256, 0, stream>>>(mW1, 96, 96, pMW12);
  pack_b_kernel<<<(3 * 6 * 512 + 255) / 256, 256, 0, stream>>>(mW2, 96, 96,
                                                               pMW12 + 18 * 512);
  pack_b_kernel<<<(3 * 6 * 512 + 255) / 256, 256, 0, stream>>>(mW3, 96, 96, pMW3);
  pack_lstm_kernel<<<(9 * 24 * 512 + 255) / 256, 256, 0, stream>>>(Wih, Whh, pG);

  (void)hipMemsetAsync(hr16, 0, NH * 2, stream);  // rnn_h0 = 0
  (void)hipMemsetAsync(m32, 0, NH * 4, stream);   // scatter accumulator
  (void)hipMemsetAsync(c32, 0, NH * 4, stream);   // rnn_c0 = 0

  input_mlp_kernel<<<NTOT, 96, 0, stream>>>(cost, rows, cols, DE, RE, CE, iW0,
                                            ib0, iW1, ib1, iW2, ib2, iW3, ib3,
                                            x16, h16);

  for (int t = 0; t < NSTEP; ++t) {
    msg_mlp_kernel<<<EDGES / 64, 256, 0, stream>>>(h16, src, dst, pMW0, pMW12,
                                                   pMW3, mb0, mb1, mb2, mb3,
                                                   m32);
    gates_gemm_kernel<<<NTOT / 64, 256, 0, stream>>>(x16, m32, hr16, pG, gates);
    lstm_update_kernel<<<(int)(NH / 256), 256, 0, stream>>>(gates, c32, hs32,
                                                            h16, hr16, m32, t);
  }

  float* outf = (float*)d_out;
  float* lossp = outf + (out_size - 1);
  zero_loss_kernel<<<1, 1, 0, stream>>>(lossp);
  const int TN = NSTEP * NTOT;  // 41472
  output_head_kernel<<<TN / 256, 256, 0, stream>>>(
      hs32, Wout, bout, labels, outf, lossp, 1.0f / (float)TN);
}
